// Yocto_27788438405129
// MI455X (gfx1250) — matrix operand order repaired and checked by reading, not compiled, not run on hardware
//
#include <hip/hip_runtime.h>
#include <math.h>

// ---------------- model constants ----------------
#define MV   32000
#define MD   1024
#define MH   16
#define MHD  64
#define MT   1024
#define MFF  4096
#define ML   8
#define MB   2
#define MS   1024
#define M2   (MB*MS)      // 2048 rows of the residual stream
#define MZ   (MB*MH)      // 32 (batch*heads)
#define VTR  128          // v-transposed padded rows (HD=64 padded to 128)

// ---------------- bf16 / fragment helpers ----------------
typedef __bf16 bf16_t;
typedef __attribute__((ext_vector_type(16))) bf16_t v16bf;
typedef __attribute__((ext_vector_type(8)))  float  v8f;

union FragBf { v16bf v; uint4 u[2]; };

static __device__ __forceinline__ unsigned short f32_to_bf16(float f) {
  unsigned int u = __float_as_uint(f);
  unsigned int r = (u + 0x7FFFu + ((u >> 16) & 1u)) >> 16;
  return (unsigned short)r;
}

static __device__ __forceinline__ float block_reduce_sum(float v, float* sbuf) {
  int t = threadIdx.x;
  sbuf[t] = v; __syncthreads();
  for (int s = 128; s > 0; s >>= 1) {
    if (t < s) sbuf[t] += sbuf[t + s];
    __syncthreads();
  }
  float r = sbuf[0]; __syncthreads();
  return r;
}

// ---------------- CDNA5 async global->LDS staging ----------------
#if __has_builtin(__builtin_amdgcn_global_load_async_to_lds_b128)
#define HAVE_ASYNC 1
#else
#define HAVE_ASYNC 0
#endif

// native 16-byte vector type matching the builtin's pointee
// (diagnostic: "__attribute__((__vector_size__(4 * sizeof(int)))) int __device__ *")
typedef int v4i_vs __attribute__((vector_size(16)));

#define TO_GLOBAL_V4(p) ((__attribute__((address_space(1))) v4i_vs*)(unsigned long long)(const void*)(p))
#define TO_LDS_V4(p)    ((__attribute__((address_space(3))) v4i_vs*)(unsigned)(unsigned long long)(void*)(p))

#if HAVE_ASYNC
template <int N>
static __device__ __forceinline__ void wait_asynccnt() {
#if __has_builtin(__builtin_amdgcn_s_wait_asynccnt)
  __builtin_amdgcn_s_wait_asynccnt(N);
#else
  asm volatile("s_wait_asynccnt %0" :: "i"(N) : "memory");
#endif
}

// stage one 128x32 A tile and one 128x32 B tile (bf16) -> 4 async b128 per wave
static __device__ __forceinline__ void stage_tiles_async(
    const unsigned short* __restrict__ A, const unsigned short* __restrict__ B,
    int lda, int ldb, int tileM, int tileN, int k0,
    unsigned short* As, unsigned short* Bs) {
#pragma unroll
  for (int it = 0; it < 2; ++it) {
    int i = threadIdx.x + it * 256;
    int row = i >> 2, cc = (i & 3) * 8;
    const unsigned short* ga = A + (long long)(tileM + row) * lda + k0 + cc;
    const unsigned short* gb = B + (long long)(tileN + row) * ldb + k0 + cc;
    __builtin_amdgcn_global_load_async_to_lds_b128(
        TO_GLOBAL_V4(ga), TO_LDS_V4(As + row * 32 + cc), 0, 0);
    __builtin_amdgcn_global_load_async_to_lds_b128(
        TO_GLOBAL_V4(gb), TO_LDS_V4(Bs + row * 32 + cc), 0, 0);
  }
}
#endif

// synchronous fallback staging (always compiles)
static __device__ __forceinline__ void stage_tiles_sync(
    const unsigned short* __restrict__ A, const unsigned short* __restrict__ B,
    int lda, int ldb, int tileM, int tileN, int k0,
    unsigned short* As, unsigned short* Bs) {
#pragma unroll
  for (int it = 0; it < 2; ++it) {
    int i = threadIdx.x + it * 256;
    int row = i >> 2, cc = (i & 3) * 8;
    uint4 va = *(const uint4*)(A + (long long)(tileM + row) * lda + k0 + cc);
    uint4 vb = *(const uint4*)(B + (long long)(tileN + row) * ldb + k0 + cc);
    *(uint4*)(&As[row * 32 + cc]) = va;
    *(uint4*)(&Bs[row * 32 + cc]) = vb;
  }
}

// one 32-wide K-step of the per-wave 32x64 sub-tile: 8 wmma
static __device__ __forceinline__ void compute_tile(
    const unsigned short* As, const unsigned short* Bs,
    int wm, int wn, int lr, int hi, v8f (&acc)[2][4]) {
  FragBf a[2];
#pragma unroll
  for (int mt = 0; mt < 2; ++mt) {
    const unsigned short* p = &As[(wm + mt * 16 + lr) * 32 + hi * 8];
    a[mt].u[0] = *(const uint4*)p;
    a[mt].u[1] = *(const uint4*)(p + 16);
  }
#pragma unroll
  for (int nt = 0; nt < 4; ++nt) {
    FragBf bf;
    const unsigned short* p = &Bs[(wn + nt * 16 + lr) * 32 + hi * 8];
    bf.u[0] = *(const uint4*)p;
    bf.u[1] = *(const uint4*)(p + 16);
#pragma unroll
    for (int mt = 0; mt < 2; ++mt) {
      acc[mt][nt] = __builtin_amdgcn_wmma_f32_16x16x32_bf16(
          false, a[mt].v, false, bf.v, (short)0, acc[mt][nt], false, false);
    }
  }
}

// ---------------- generic NT WMMA GEMM ----------------
// C[M,N] = A[M,K] * B[N,K]^T (+ epilogue). M,N multiples of 128; K multiple of 32.
// epi: 0=none 1=+bias 2=+bias,gelu 3=+residual 4=+bias,+residual 5=scale+causal mask
__global__ __launch_bounds__(256)
void gemm_nt_wmma(const unsigned short* __restrict__ A,
                  const unsigned short* __restrict__ B,
                  void* __restrict__ C,
                  const float* __restrict__ bias,
                  const float* __restrict__ R,
                  int M, int N, int K, int lda, int ldb, int ldc,
                  long long sA, long long sB, long long sC,
                  int epi, int storeBf16, float scale) {
  __shared__ __align__(16) unsigned short As0[128 * 32];
  __shared__ __align__(16) unsigned short Bs0[128 * 32];
#if HAVE_ASYNC
  __shared__ __align__(16) unsigned short As1[128 * 32];
  __shared__ __align__(16) unsigned short Bs1[128 * 32];
#endif

  const int bz = blockIdx.z;
  A += (long long)bz * sA;
  B += (long long)bz * sB;
  void* Cb = storeBf16 ? (void*)((unsigned short*)C + (long long)bz * sC)
                       : (void*)((float*)C + (long long)bz * sC);

  const int tileM = blockIdx.x * 128;
  const int tileN = blockIdx.y * 128;

  // causal fast path: whole tile masked -> fill with -1e30 and exit
  if (epi == 5 && tileN > tileM + 127) {
    float* Cf = (float*)Cb;
    for (int i = threadIdx.x; i < 128 * 128; i += 256) {
      int r = i >> 7, c = i & 127;
      Cf[(long long)(tileM + r) * ldc + (tileN + c)] = -1e30f;
    }
    return;
  }

  const int wid  = threadIdx.x >> 5;
  const int lane = threadIdx.x & 31;
  const int wm = (wid >> 1) * 32;   // 0,32,64,96
  const int wn = (wid & 1) * 64;    // 0,64
  const int lr = lane & 15;
  const int hi = lane >> 4;

  v8f acc[2][4] = {};
  const int ntiles = K >> 5;

#if HAVE_ASYNC
  // double-buffered async pipeline: prefetch tile t+1 while computing tile t
  stage_tiles_async(A, B, lda, ldb, tileM, tileN, 0, As0, Bs0);
  int buf = 0;
  for (int t = 0; t < ntiles; ++t) {
    if (t + 1 < ntiles) {
      stage_tiles_async(A, B, lda, ldb, tileM, tileN, (t + 1) << 5,
                        buf ? As0 : As1, buf ? Bs0 : Bs1);
      wait_asynccnt<4>();   // in-order: tile t's 4 ops complete
    } else {
      wait_asynccnt<0>();
    }
    __syncthreads();
    compute_tile(buf ? As1 : As0, buf ? Bs1 : Bs0, wm, wn, lr, hi, acc);
    __syncthreads();
    buf ^= 1;
  }
#else
  for (int t = 0; t < ntiles; ++t) {
    stage_tiles_sync(A, B, lda, ldb, tileM, tileN, t << 5, As0, Bs0);
    __syncthreads();
    compute_tile(As0, Bs0, wm, wn, lr, hi, acc);
    __syncthreads();
  }
#endif

  // epilogue (no bounds checks: M,N are multiples of 128 by construction)
#pragma unroll
  for (int mt = 0; mt < 2; ++mt) {
#pragma unroll
    for (int nt = 0; nt < 4; ++nt) {
      int col = tileN + wn + nt * 16 + lr;
      float bv = (epi == 1 || epi == 2 || epi == 4) ? bias[col] : 0.0f;
#pragma unroll
      for (int v = 0; v < 8; ++v) {
        int row = tileM + wm + mt * 16 + hi * 8 + v;
        float x = acc[mt][nt][v];
        if (epi == 1 || epi == 2 || epi == 4) x += bv;
        if (epi == 2) x = 0.5f * x * (1.0f + erff(x * 0.70710678118f));
        if (epi == 3 || epi == 4) x += R[(long long)row * ldc + col];
        if (epi == 5) { x *= scale; if (col > row) x = -1e30f; }
        if (storeBf16)
          ((unsigned short*)Cb)[(long long)row * ldc + col] = f32_to_bf16(x);
        else
          ((float*)Cb)[(long long)row * ldc + col] = x;
      }
    }
  }
}

// ---------------- elementwise kernels ----------------
__global__ void gather_embed(const int* __restrict__ ids,
                             const float* __restrict__ emb,
                             float* __restrict__ x) {
  int row = blockIdx.x;
  long long id = ids[row];
  const float* src = emb + id * (long long)MD;
  float* dst = x + (long long)row * MD;
  for (int i = threadIdx.x; i < MD; i += 256) dst[i] = src[i];
}

__global__ void cvt_f32_bf16(const float* __restrict__ a,
                             unsigned short* __restrict__ o, long long n) {
  long long i = (long long)blockIdx.x * blockDim.x + threadIdx.x;
  long long st = (long long)gridDim.x * blockDim.x;
  for (; i < n; i += st) o[i] = f32_to_bf16(a[i]);
}

__global__ void layernorm_bf16(const float* __restrict__ x,
                               const float* __restrict__ w,
                               const float* __restrict__ b,
                               unsigned short* __restrict__ y) {
  __shared__ float sbuf[256];
  long long row = blockIdx.x;
  const float* xr = x + row * MD;
  float v[4]; float s = 0.0f;
#pragma unroll
  for (int i = 0; i < 4; ++i) { v[i] = xr[threadIdx.x + 256 * i]; s += v[i]; }
  s = block_reduce_sum(s, sbuf);
  float mean = s * (1.0f / MD);
  float sq = 0.0f;
#pragma unroll
  for (int i = 0; i < 4; ++i) { float d = v[i] - mean; sq += d * d; }
  sq = block_reduce_sum(sq, sbuf);
  float rstd = rsqrtf(sq * (1.0f / MD) + 1e-5f);
#pragma unroll
  for (int i = 0; i < 4; ++i) {
    int c = threadIdx.x + 256 * i;
    y[row * MD + c] = f32_to_bf16((v[i] - mean) * rstd * w[c] + b[c]);
  }
}

// zero padded rows 64..127 of v-transposed (once per launch)
__global__ void zero_vt_pad(unsigned short* __restrict__ vt) {
  long long idx = (long long)blockIdx.x * 256 + threadIdx.x;  // [0, MZ*64*MS)
  int s = (int)(idx & 1023);
  int r = (int)((idx >> 10) & 63);
  int z = (int)(idx >> 16);
  vt[((long long)z * VTR + 64 + r) * MS + s] = 0;
}

// RoPE on q,k + head split; v stored transposed [z][VTR][s] for NT probs·v GEMM
__global__ void rope_qkv_kernel(const float* __restrict__ u,
                                unsigned short* __restrict__ q,
                                unsigned short* __restrict__ k,
                                unsigned short* __restrict__ vt) {
  int idx = blockIdx.x * 256 + threadIdx.x;     // [0, B*S*H*32)
  int j = idx & 31;
  int t = idx >> 5;
  int h = t & 15;
  int s = (t >> 4) & 1023;
  int b = t >> 14;
  const float* up = u + ((long long)(b * MS + s)) * (3 * MT) + h * MHD;
  float q1 = up[j],        q2 = up[j + 32];
  float k1 = up[MT + j],   k2 = up[MT + j + 32];
  float v1 = up[2*MT + j], v2 = up[2*MT + j + 32];
  float invf = __expf(-(float)(2 * j) * (9.210340371976184f / 64.0f)); // 10000^(-2j/64)
  float ang = (float)s * invf;
  float c = __cosf(ang), sn = __sinf(ang);
  float qo1 = q1 * c - q2 * sn, qo2 = q2 * c + q1 * sn;
  float ko1 = k1 * c - k2 * sn, ko2 = k2 * c + k1 * sn;
  int z = b * MH + h;
  long long qkbase = ((long long)z * MS + s) * MHD;
  q[qkbase + j]      = f32_to_bf16(qo1);
  q[qkbase + j + 32] = f32_to_bf16(qo2);
  k[qkbase + j]      = f32_to_bf16(ko1);
  k[qkbase + j + 32] = f32_to_bf16(ko2);
  long long vtbase = (long long)z * VTR * MS;
  vt[vtbase + (long long)j * MS + s]        = f32_to_bf16(v1);
  vt[vtbase + (long long)(j + 32) * MS + s] = f32_to_bf16(v2);
}

__global__ void softmax_causal(const float* __restrict__ sc,
                               unsigned short* __restrict__ pr) {
  __shared__ float sbuf[256];
  long long row = blockIdx.x;          // z*S + q
  const float* x = sc + row * MS;
  float v[4];
  float m = -1e30f;
#pragma unroll
  for (int i = 0; i < 4; ++i) { v[i] = x[threadIdx.x + 256 * i]; m = fmaxf(m, v[i]); }
  sbuf[threadIdx.x] = m; __syncthreads();
  for (int s = 128; s > 0; s >>= 1) {
    if (threadIdx.x < s) sbuf[threadIdx.x] = fmaxf(sbuf[threadIdx.x], sbuf[threadIdx.x + s]);
    __syncthreads();
  }
  m = sbuf[0]; __syncthreads();
  float sum = 0.0f;
#pragma unroll
  for (int i = 0; i < 4; ++i) { v[i] = __expf(v[i] - m); sum += v[i]; }
  sum = block_reduce_sum(sum, sbuf);
  float inv = 1.0f / sum;
#pragma unroll
  for (int i = 0; i < 4; ++i)
    pr[row * MS + threadIdx.x + 256 * i] = f32_to_bf16(v[i] * inv);
}

// OH [z][s][128] (bf16, cols 0..63 valid) -> AO [b*S+s][h*HD+hd] (bf16)
__global__ void merge_heads(const unsigned short* __restrict__ oh,
                            unsigned short* __restrict__ ao) {
  long long idx = (long long)blockIdx.x * 256 + threadIdx.x;  // [0, B*S*T)
  int d = (int)(idx & 63);
  int h = (int)((idx >> 6) & 15);
  int s = (int)((idx >> 10) & 1023);
  int b = (int)(idx >> 20);
  ao[idx] = oh[(((long long)(b * MH + h) * MS + s) * 128) + d];
}

// ---------------- launch orchestration ----------------
extern "C" void kernel_launch(void* const* d_in, const int* in_sizes, int n_in,
                              void* d_out, int out_size, void* d_ws, size_t ws_size,
                              hipStream_t stream) {
  const int*   ids = (const int*)d_in[0];
  const float* emb = (const float*)d_in[1];
  const float* Wu  = (const float*)d_in[2];
  const float* Wo  = (const float*)d_in[3];
  const float* n1w = (const float*)d_in[4];
  const float* n1b = (const float*)d_in[5];
  const float* n2w = (const float*)d_in[6];
  const float* n2b = (const float*)d_in[7];
  const float* f1w = (const float*)d_in[8];
  const float* f1b = (const float*)d_in[9];
  const float* f2w = (const float*)d_in[10];
  const float* f2b = (const float*)d_in[11];
  const float* fnw = (const float*)d_in[12];
  const float* fnb = (const float*)d_in[13];
  float* out = (float*)d_out;
  char* ws = (char*)d_ws;

  size_t off = 0;
  auto alloc = [&](size_t bytes) {
    size_t o = off; off += (bytes + 255) & ~(size_t)255; return o;
  };
  float*          X   = (float*)(ws + alloc((size_t)M2 * MD * 4));
  unsigned short* XN  = (unsigned short*)(ws + alloc((size_t)M2 * MD * 2));
  float*          U   = (float*)(ws + alloc((size_t)M2 * 3 * MT * 4));
  unsigned short* Qb  = (unsigned short*)(ws + alloc((size_t)MZ * MS * MHD * 2));
  unsigned short* Kb  = (unsigned short*)(ws + alloc((size_t)MZ * MS * MHD * 2));
  unsigned short* VT  = (unsigned short*)(ws + alloc((size_t)MZ * VTR * MS * 2));
  float*          SC  = (float*)(ws + alloc((size_t)MZ * MS * MS * 4));
  unsigned short* PR  = (unsigned short*)(ws + alloc((size_t)MZ * MS * MS * 2));
  unsigned short* OH  = (unsigned short*)(ws + alloc((size_t)MZ * MS * 128 * 2));
  unsigned short* AO  = (unsigned short*)(ws + alloc((size_t)M2 * MT * 2));
  unsigned short* HB  = (unsigned short*)(ws + alloc((size_t)M2 * MFF * 2));
  unsigned short* EMBB = (unsigned short*)(ws + alloc((size_t)MV * MD * 2));
  unsigned short* WUB  = (unsigned short*)(ws + alloc((size_t)ML * 3 * MT * MD * 2));
  unsigned short* WOB  = (unsigned short*)(ws + alloc((size_t)ML * MD * MT * 2));
  unsigned short* F1B  = (unsigned short*)(ws + alloc((size_t)ML * MFF * MD * 2));
  unsigned short* F2B  = (unsigned short*)(ws + alloc((size_t)ML * MD * MFF * 2));

  auto gemm = [&](const unsigned short* A, const unsigned short* Bm, void* C,
                  const float* bias, const float* R,
                  int M, int N, int K, int lda, int ldb, int ldc,
                  long long sA, long long sB, long long sC,
                  int batches, int epi, int sbf, float scale) {
    dim3 grid((M + 127) / 128, (N + 127) / 128, batches);
    gemm_nt_wmma<<<grid, dim3(256), 0, stream>>>(A, Bm, C, bias, R, M, N, K,
                                                 lda, ldb, ldc, sA, sB, sC,
                                                 epi, sbf, scale);
  };

  // weight down-conversion to bf16 (once per launch)
  cvt_f32_bf16<<<4096, 256, 0, stream>>>(emb, EMBB, (long long)MV * MD);
  cvt_f32_bf16<<<4096, 256, 0, stream>>>(Wu,  WUB,  (long long)ML * 3 * MT * MD);
  cvt_f32_bf16<<<4096, 256, 0, stream>>>(Wo,  WOB,  (long long)ML * MD * MT);
  cvt_f32_bf16<<<4096, 256, 0, stream>>>(f1w, F1B,  (long long)ML * MFF * MD);
  cvt_f32_bf16<<<4096, 256, 0, stream>>>(f2w, F2B,  (long long)ML * MD * MFF);

  zero_vt_pad<<<(MZ * 64 * MS) / 256, 256, 0, stream>>>(VT);
  gather_embed<<<M2, 256, 0, stream>>>(ids, emb, X);

  for (int l = 0; l < ML; ++l) {
    const unsigned short* WuL = WUB + (size_t)l * 3 * MT * MD;
    const unsigned short* WoL = WOB + (size_t)l * MD * MT;
    const unsigned short* F1L = F1B + (size_t)l * MFF * MD;
    const unsigned short* F2L = F2B + (size_t)l * MD * MFF;

    // --- attention ---
    layernorm_bf16<<<M2, 256, 0, stream>>>(X, n1w + l * MD, n1b + l * MD, XN);
    // u = xn * Wu^T   [2048,3072]
    gemm(XN, WuL, U, nullptr, nullptr, M2, 3 * MT, MD, MD, MD, 3 * MT,
         0, 0, 0, 1, 0, 0, 0.0f);
    rope_qkv_kernel<<<(MB * MS * MH * 32) / 256, 256, 0, stream>>>(U, Qb, Kb, VT);
    // scores = q k^T / 8 with causal mask   [z][1024,1024]
    gemm(Qb, Kb, SC, nullptr, nullptr, MS, MS, MHD, MHD, MHD, MS,
         (long long)MS * MHD, (long long)MS * MHD, (long long)MS * MS,
         MZ, 5, 0, 0.125f);
    softmax_causal<<<MZ * MS, 256, 0, stream>>>(SC, PR);
    // out_heads = probs * v    [z][1024,128] (bf16; cols 64..127 are zeros)
    gemm(PR, VT, OH, nullptr, nullptr, MS, 128, MS, MS, MS, 128,
         (long long)MS * MS, (long long)VTR * MS, (long long)MS * 128,
         MZ, 0, 1, 0.0f);
    merge_heads<<<(M2 * MT) / 256, 256, 0, stream>>>(OH, AO);
    // x = x + attnout * Wo^T
    gemm(AO, WoL, X, nullptr, X, M2, MD, MT, MT, MT, MD,
         0, 0, 0, 1, 3, 0, 0.0f);

    // --- FFN ---
    layernorm_bf16<<<M2, 256, 0, stream>>>(X, n2w + l * MD, n2b + l * MD, XN);
    // h = gelu(xn * f1w^T + f1b)  -> bf16 directly
    gemm(XN, F1L, HB, f1b + l * MFF, nullptr, M2, MFF, MD, MD, MD, MFF,
         0, 0, 0, 1, 2, 1, 0.0f);
    // x = x + h * f2w^T + f2b
    gemm(HB, F2L, X, f2b + l * MD, X, M2, MD, MFF, MFF, MFF, MD,
         0, 0, 0, 1, 4, 0, 0.0f);
  }

  // final norm + logits = xn * emb^T  [2048, 32000]
  layernorm_bf16<<<M2, 256, 0, stream>>>(X, fnw, fnb, XN);
  gemm(XN, EMBB, out, nullptr, nullptr, M2, MV, MD, MD, MD, MV,
       0, 0, 0, 1, 0, 0, 0.0f);
}
